// MHLocAttention_19207093748545
// MI455X (gfx1250) — matrix operand order repaired and checked by reading, not compiled, not run on hardware
//
#include <hip/hip_runtime.h>
#include <cstdint>
#include <cstddef>

// dims
#define N_   16
#define T_   2000
#define ENC_ 512
#define DEC_ 512
#define ATT_ 512
#define C_   10
#define KC_  64          // conv half width
#define KW_  129         // 2*KC_+1
#define H_   4
#define GR_  (ATT_*H_)   // 2048 projected rows (H*ATT)
#define TT_  64          // t tile per workgroup
#define LDSROW 520       // LDS row stride in bf16 elems (512 + pad, 16B aligned)

typedef __attribute__((ext_vector_type(16))) __bf16        v16bf;
typedef __attribute__((ext_vector_type(8)))  float         v8f;
typedef __attribute__((ext_vector_type(8)))  unsigned int  v8u;

__device__ __forceinline__ unsigned short f2bf(float f) {
  unsigned u = __builtin_bit_cast(unsigned, f);
  u += 0x7FFFu + ((u >> 16) & 1u);          // round-to-nearest-even
  return (unsigned short)(u >> 16);
}
__device__ __forceinline__ float fast_exp(float x) {
  return __builtin_amdgcn_exp2f(x * 1.44269504f);            // v_exp_f32
}
__device__ __forceinline__ float fast_tanh(float x) {
  float e = __builtin_amdgcn_exp2f(x * 2.885390082f);        // e^(2x)
  return 1.0f - 2.0f * __builtin_amdgcn_rcpf(e + 1.0f);
}

// ---- prep kernels -----------------------------------------------------------

__global__ void k_prep_wk(const float* __restrict__ Wk, unsigned short* __restrict__ out) {
  int i = blockIdx.x * 256 + threadIdx.x;   // exactly GR_*ENC_ threads
  out[i] = f2bf(Wk[i]);
}

__global__ void k_prep_dec(const float* __restrict__ dec, const float* __restrict__ Wd,
                           float* __restrict__ decp) {
  int idx = blockIdx.x * 256 + threadIdx.x;          // N_*GR_
  int j = idx & (GR_ - 1), n = idx >> 11;
  const float* w = Wd + (size_t)j * DEC_;
  const float* x = dec + (size_t)n * DEC_;
  float a = 0.f;
  for (int e = 0; e < DEC_; ++e) a += x[e] * w[e];
  decp[idx] = a;
}

__global__ void k_conv(const float* __restrict__ ali_prev, const float* __restrict__ WF,
                       const float* __restrict__ bF, float* __restrict__ fbuf) {
  int idx = blockIdx.x * 256 + threadIdx.x;          // N_*H_*C_*T_
  if (idx >= N_ * H_ * C_ * T_) return;
  int t = idx % T_; int q = idx / T_;
  int c = q % C_;  q /= C_;
  int h = q % H_;  int n = q / H_;
  int ch = h * C_ + c;
  const float* w = WF + (size_t)ch * KW_;
  const float* x = ali_prev + ((size_t)n * H_ + h) * T_;
  float a = bF[ch];
  for (int j = 0; j < KW_; ++j) {
    int tt = t + j - KC_;
    if (tt >= 0 && tt < T_) a += w[j] * x[tt];
  }
  fbuf[idx] = a;                                     // [n][h][c][t]
}

// ---- fused score GEMM: key_part + att_part + dec_part -> tanh -> Ww·reduce --

__global__ __launch_bounds__(256) void k_score(
    const float* __restrict__ enc,                 // N,T,ENC
    const unsigned short* __restrict__ WkB,        // GR_ x ENC_ bf16
    const float* __restrict__ decp,                // N x GR_
    const float* __restrict__ fbuf,                // N,H,C,T
    const float* __restrict__ Wa,                  // H,ATT,C
    const float* __restrict__ Ww,                  // H,ATT
    float* __restrict__ score)                     // N,H,T
{
  __shared__ unsigned short ldsB[TT_ * LDSROW];    // enc tile, bf16 [t][e]
  __shared__ float sAcc[4 * H_ * TT_];             // 4 contributors x H x 64

  const int tid = threadIdx.x;
  const int n   = blockIdx.y;
  const int t0  = blockIdx.x * TT_;

  // Stage enc tile to LDS as bf16 (guarded at T edge).
  for (int i = tid; i < TT_ * (ENC_ / 4); i += 256) {
    int r = i >> 7, c4 = i & 127;
    int t = t0 + r;
    float4 v = make_float4(0.f, 0.f, 0.f, 0.f);
    if (t < T_) v = *reinterpret_cast<const float4*>(enc + ((size_t)n * T_ + t) * ENC_ + c4 * 4);
    unsigned lo = (unsigned)f2bf(v.x) | ((unsigned)f2bf(v.y) << 16);
    unsigned hi = (unsigned)f2bf(v.z) | ((unsigned)f2bf(v.w) << 16);
    *reinterpret_cast<uint2*>(&ldsB[r * LDSROW + c4 * 4]) = make_uint2(lo, hi);
  }
  __syncthreads();

  const int lane = tid & 31;
  const int wave = tid >> 5;
  const int hf   = lane >> 4;       // lane half (WMMA operand striping)
  const int l16  = lane & 15;
  const int tsub = wave & 3;        // which 16-col slab of the 64-wide tile
  const int doff = wave >> 2;       // d-tile parity (2 waves per t slab)
  const int tloc = tsub * 16 + l16; // this lane's column within the tile
  const int tglob = t0 + tloc;
  const bool tvalid = (tglob < T_);

  const unsigned short* brow = &ldsB[tloc * LDSROW];
  const float* fcol = fbuf + (size_t)n * H_ * C_ * T_ + (tvalid ? tglob : 0);

  // Single long d-tile loop (best A-load pipelining); per-head accumulation
  // via compare-select so everything stays in registers.
  float acc0 = 0.f, acc1 = 0.f, acc2 = 0.f, acc3 = 0.f;

  for (int dti = 0; dti < 64; ++dti) {
    const int dtile = doff + dti * 2;              // 0..127 across the wave pair
    const int row0  = dtile * 16;                  // global projected row
    const int h     = dtile >> 5;                  // head (runtime, address-only)
    const unsigned short* arow = WkB + (size_t)(row0 + l16) * ENC_;

    v8f acc = {0.f, 0.f, 0.f, 0.f, 0.f, 0.f, 0.f, 0.f};
    #pragma unroll
    for (int k0 = 0; k0 < ENC_; k0 += 32) {
      // A fragment (Wk rows): two contiguous 16B runs per lane
      uint4 a0 = *reinterpret_cast<const uint4*>(arow + k0 + hf * 8);
      uint4 a1 = *reinterpret_cast<const uint4*>(arow + k0 + 16 + hf * 8);
      // B fragment (enc^T cols): two contiguous 16B runs per lane from LDS
      // (loop-invariant across d-tiles; compiler hoists into registers)
      uint4 b0 = *reinterpret_cast<const uint4*>(brow + k0 + hf * 8);
      uint4 b1 = *reinterpret_cast<const uint4*>(brow + k0 + 16 + hf * 8);
      v8u au = {a0.x, a0.y, a0.z, a0.w, a1.x, a1.y, a1.z, a1.w};
      v8u bu = {b0.x, b0.y, b0.z, b0.w, b1.x, b1.y, b1.z, b1.w};
      acc = __builtin_amdgcn_wmma_f32_16x16x32_bf16(
          false, __builtin_bit_cast(v16bf, au),
          false, __builtin_bit_cast(v16bf, bu),
          (short)0, acc, false, false);
    }

    // Location-conv features for this (head, column): constant-indexed array
    // (registers); dynamic h only affects the load address. L0-hot.
    float fc[C_];
    #pragma unroll
    for (int c = 0; c < C_; ++c) {
      float v = fcol[((size_t)h * C_ + c) * T_];
      fc[c] = tvalid ? v : 0.f;
    }

    // Epilogue: + dec_part + att_part, tanh, Ww-weight, reduce 8 rows.
    float s = 0.f;
    #pragma unroll
    for (int i = 0; i < 8; ++i) {
      int m = hf * 8 + i;                       // row within tile (C/D layout)
      int d = (row0 & (ATT_ - 1)) + m;          // row within head
      float v = acc[i] + decp[(size_t)n * GR_ + row0 + m];
      const float* wa = Wa + ((size_t)h * ATT_ + d) * C_;
      float attv = 0.f;
      #pragma unroll
      for (int c = 0; c < C_; ++c) attv += wa[c] * fc[c];
      s += Ww[h * ATT_ + d] * fast_tanh(v + attv);
    }
    acc0 += (h == 0) ? s : 0.f;
    acc1 += (h == 1) ? s : 0.f;
    acc2 += (h == 2) ? s : 0.f;
    acc3 += (h == 3) ? s : 0.f;
  }

  // Deterministic cross-wave/half reduction: distinct slot per contributor.
  const int contrib = doff * 2 + hf;           // 0..3
  sAcc[(contrib * H_ + 0) * TT_ + tloc] = acc0;
  sAcc[(contrib * H_ + 1) * TT_ + tloc] = acc1;
  sAcc[(contrib * H_ + 2) * TT_ + tloc] = acc2;
  sAcc[(contrib * H_ + 3) * TT_ + tloc] = acc3;
  __syncthreads();

  {
    int h = tid >> 6, col = tid & 63, t = t0 + col;
    if (t < T_) {
      float s = 0.f;
      #pragma unroll
      for (int c = 0; c < 4; ++c) s += sAcc[(c * H_ + h) * TT_ + col];
      score[((size_t)n * H_ + h) * T_ + t] = s;
    }
  }
}

// ---- masked softmax over T --------------------------------------------------

__global__ __launch_bounds__(256) void k_softmax(
    const float* __restrict__ score, const int* __restrict__ enc_len,
    float* __restrict__ ali) {
  __shared__ float red[256];
  int nh = blockIdx.x, n = nh >> 2, tid = threadIdx.x;
  int L = enc_len[n];
  const float* s = score + (size_t)nh * T_;
  float* o = ali + (size_t)nh * T_;

  float mx = -3.4e38f;
  for (int t = tid; t < L; t += 256) mx = fmaxf(mx, s[t]);
  red[tid] = mx; __syncthreads();
  for (int st = 128; st > 0; st >>= 1) { if (tid < st) red[tid] = fmaxf(red[tid], red[tid + st]); __syncthreads(); }
  mx = red[0]; __syncthreads();

  float sum = 0.f;
  for (int t = tid; t < L; t += 256) { float e = fast_exp(s[t] - mx); o[t] = e; sum += e; }
  red[tid] = sum; __syncthreads();
  for (int st = 128; st > 0; st >>= 1) { if (tid < st) red[tid] += red[tid + st]; __syncthreads(); }
  float inv = 1.0f / red[0];

  for (int t = tid; t < L; t += 256) o[t] *= inv;
  for (int t = L + tid; t < T_; t += 256) o[t] = 0.f;   // masked -> 0
}

// ---- attend over raw encoder, then project (algebraic reassociation) --------

__global__ void k_wenc(const float* __restrict__ ali, const float* __restrict__ enc,
                       const int* __restrict__ enc_len, float* __restrict__ wenc) {
  int idx = blockIdx.x * 256 + threadIdx.x;      // N_*H_*ENC_
  int e = idx & (ENC_ - 1);
  int nh = idx >> 9, n = nh >> 2;
  int L = enc_len[n];
  const float* a = ali + (size_t)nh * T_;
  const float* ep = enc + (size_t)n * T_ * ENC_ + e;
  float acc = 0.f;
  for (int t = 0; t < L; ++t) acc += a[t] * ep[(size_t)t * ENC_];
  wenc[idx] = acc;                               // [n][h][e]
}

__global__ void k_proj1(const float* __restrict__ wenc, const float* __restrict__ We,
                        const float* __restrict__ be, float* __restrict__ ctxp) {
  int idx = blockIdx.x * 256 + threadIdx.x;      // N_*GR_
  int j = idx & (GR_ - 1), n = idx >> 11, h = j >> 9;
  const float* w = We + (size_t)j * ENC_;
  const float* x = wenc + ((size_t)n * H_ + h) * ENC_;
  float acc = be[j];
  for (int e = 0; e < ENC_; ++e) acc += x[e] * w[e];
  ctxp[idx] = acc;
}

__global__ void k_proj2(const float* __restrict__ ctxp, const float* __restrict__ Wc,
                        const float* __restrict__ bc, float* __restrict__ out) {
  int idx = blockIdx.x * 256 + threadIdx.x;      // N_*ENC_
  int o = idx & (ENC_ - 1), n = idx >> 9;
  const float* w = Wc + (size_t)o * GR_;
  const float* x = ctxp + (size_t)n * GR_;
  float acc = bc[o];
  for (int j = 0; j < GR_; ++j) acc += x[j] * w[j];
  out[idx] = acc;
}

// ---- launcher ---------------------------------------------------------------

extern "C" void kernel_launch(void* const* d_in, const int* in_sizes, int n_in,
                              void* d_out, int out_size, void* d_ws, size_t ws_size,
                              hipStream_t stream) {
  (void)in_sizes; (void)n_in; (void)out_size; (void)ws_size;
  const float* enc     = (const float*)d_in[0];
  const int*   enc_len = (const int*)  d_in[1];
  const float* dec     = (const float*)d_in[2];
  const float* ali_p   = (const float*)d_in[3];
  const float* We      = (const float*)d_in[4];
  const float* be      = (const float*)d_in[5];
  const float* Wk      = (const float*)d_in[6];
  const float* Wd      = (const float*)d_in[7];
  const float* WF      = (const float*)d_in[8];
  const float* bF      = (const float*)d_in[9];
  const float* Wa      = (const float*)d_in[10];
  const float* Ww      = (const float*)d_in[11];
  const float* Wc      = (const float*)d_in[12];
  const float* bc      = (const float*)d_in[13];

  float* out     = (float*)d_out;
  float* ali_out = out;                          // N*H*T
  float* ctx_out = out + (size_t)N_ * H_ * T_;   // N*ENC

  char* ws = (char*)d_ws;
  size_t off = 0;
  auto alloc = [&](size_t bytes) { void* p = ws + off; off = (off + bytes + 255) & ~(size_t)255; return p; };
  unsigned short* WkB  = (unsigned short*)alloc((size_t)GR_ * ENC_ * sizeof(unsigned short)); // 2 MB
  float*          decp = (float*)alloc((size_t)N_ * GR_ * sizeof(float));
  float*          fbuf = (float*)alloc((size_t)N_ * H_ * C_ * T_ * sizeof(float));            // 5 MB
  float*          scor = (float*)alloc((size_t)N_ * H_ * T_ * sizeof(float));
  float*          wenc = (float*)alloc((size_t)N_ * H_ * ENC_ * sizeof(float));
  float*          ctxp = (float*)alloc((size_t)N_ * GR_ * sizeof(float));

  k_prep_wk <<<dim3(GR_ * ENC_ / 256), dim3(256), 0, stream>>>(Wk, WkB);
  k_prep_dec<<<dim3(N_ * GR_ / 256),   dim3(256), 0, stream>>>(dec, Wd, decp);
  k_conv    <<<dim3((N_ * H_ * C_ * T_ + 255) / 256), dim3(256), 0, stream>>>(ali_p, WF, bF, fbuf);
  k_score   <<<dim3((T_ + TT_ - 1) / TT_, N_), dim3(256), 0, stream>>>(enc, WkB, decp, fbuf, Wa, Ww, scor);
  k_softmax <<<dim3(N_ * H_), dim3(256), 0, stream>>>(scor, enc_len, ali_out);
  k_wenc    <<<dim3(N_ * H_ * ENC_ / 256), dim3(256), 0, stream>>>(ali_out, enc, enc_len, wenc);
  k_proj1   <<<dim3(N_ * GR_ / 256), dim3(256), 0, stream>>>(wenc, We, be, ctxp);
  k_proj2   <<<dim3(N_ * ENC_ / 256), dim3(256), 0, stream>>>(ctxp, Wc, bc, ctx_out);
}
